// KMQuantizer_24343874633975
// MI455X (gfx1250) — matrix operand order repaired and checked by reading, not compiled, not run on hardware
//
#include <hip/hip_runtime.h>
#include <hip/hip_bf16.h>

// ---------------------------------------------------------------------------
// VQ codebook quantizer for MI455X (gfx1250, wave32, WMMA + async LDS copy).
//   token   : [B=8, T=2048, D=768] f32   (d_in[0])
//   codebook: [K=2048, D=768]      f32   (d_in[1])
// Outputs (concatenated in d_out):
//   quantized [M=16384, 768] f32, indices [16384] i32, commit_loss [1] f32
// ---------------------------------------------------------------------------

#define Mtot 16384      // B*T rows
#define Dm   768        // feature dim (WMMA K)
#define Kc   2048       // codebook entries (WMMA N)
#define NSPLIT 2        // codebook partitions for extra block parallelism
#define INV_N (1.0f / (float)(Mtot * Dm))

#define KPANEL 128                      // K-elements staged per LDS panel
#define PANEL_BYTES (128 * KPANEL * 2)  // 128 cols * 128 K * bf16 = 32 KB

typedef __attribute__((ext_vector_type(16))) __bf16 v16bf;
typedef __attribute__((ext_vector_type(8)))  float  v8f;
typedef __attribute__((ext_vector_type(4)))  int    v4i;

union FragU { v4i i[2]; v16bf b; };

// ds_load_b128 pair into a fragment, LDS byte offset as an immediate.
#define DS_LOAD_FRAG(f, addr, off)                                   \
  asm volatile("ds_load_b128 %0, %2 offset:%c3\n\t"                  \
               "ds_load_b128 %1, %2 offset:%c4"                      \
               : "=v"((f).i[0]), "=v"((f).i[1])                      \
               : "v"(addr), "n"(off), "n"((off) + 16))

// s_wait_dscnt tied to the fragment so load -> wait -> WMMA order is forced.
#define DS_WAIT_FRAG(f, imm)                                         \
  asm volatile("s_wait_dscnt " imm                                   \
               : "+v"((f).i[0]), "+v"((f).i[1]))

// ---------------------------------------------------------------------------
// Kernel 1: f32 -> bf16 conversion of token and codebook into workspace.
// ---------------------------------------------------------------------------
__global__ void vq_convert(const float* __restrict__ tok,
                           const float* __restrict__ cb,
                           __bf16* __restrict__ tokbf,
                           __bf16* __restrict__ cbbf) {
  const int ntok   = Mtot * Dm;
  const int ntotal = ntok + Kc * Dm;
  for (int i = blockIdx.x * blockDim.x + threadIdx.x; i < ntotal;
       i += gridDim.x * blockDim.x) {
    if (i < ntok) tokbf[i] = (__bf16)tok[i];
    else          cbbf[i - ntok] = (__bf16)cb[i - ntok];
  }
}

// ---------------------------------------------------------------------------
// Kernel 2: e_sq[k] = sum_d codebook[k,d]^2 (f32); block 0 zeroes the loss.
// ---------------------------------------------------------------------------
__global__ void vq_esq(const float* __restrict__ cb,
                       float* __restrict__ esq,
                       float* __restrict__ out_loss) {
  const int k = blockIdx.x;
  float s = 0.f;
  for (int d = threadIdx.x; d < Dm; d += blockDim.x) {
    float v = cb[(size_t)k * Dm + d];
    s += v * v;
  }
  __shared__ float red[256];
  red[threadIdx.x] = s;
  __syncthreads();
  for (int o = 128; o > 0; o >>= 1) {
    if ((int)threadIdx.x < o) red[threadIdx.x] += red[threadIdx.x + o];
    __syncthreads();
  }
  if (threadIdx.x == 0) {
    esq[k] = red[0];
    if (k == 0) *out_loss = 0.f;   // d_out is poisoned; stream order protects us
  }
}

// ---------------------------------------------------------------------------
// Kernel 3: WMMA distance + running argmin.
// Grid = (Mtot/128) * NSPLIT blocks, 256 threads (8 waves).
// Wave w owns rows [mtile*128 + w*16); block partition p scans its half of the
// codebook in 128-column chunks.
//
// B staging: per 128-K panel, all 256 threads issue global_load_async_to_lds
// (ASYNCcnt) into a double-buffered 32 KB LDS panel laid out in WMMA fragment
// order [kstep 0..3][tile 0..7][lane 0..31][32B]; each wave's B fragment is
// two conflict-free ds_load_b128. The 32-WMMA inner loop is a hand-pinned
// depth-3 asm pipeline over 4 fragment slots: issue loads for j+3, then an
// s_wait_dscnt tied to slot j's registers before its WMMA, so 3 LDS fragment
// pairs stay in flight ahead of the matrix pipe.  A fragments are loaded
// straight from global (hoisted per panel, latency hidden by the panel wait),
// in contiguous k = 16*(lane/16) + s order, same as B.  C layout: lane = column N (mod 16),
// VGPR i = row (i or 8+i).
// ---------------------------------------------------------------------------
__global__ void __launch_bounds__(256)
vq_argmin(const __bf16* __restrict__ tokbf,
          const __bf16* __restrict__ cbbf,
          const float*  __restrict__ esq,
          float* __restrict__ bdist,
          int*   __restrict__ bidx) {
  __shared__ __attribute__((aligned(16))) unsigned char sB[2 * PANEL_BYTES];

  const int part   = blockIdx.x & (NSPLIT - 1);
  const int mtile  = blockIdx.x / NSPLIT;
  const int wave   = threadIdx.x >> 5;
  const int lane   = threadIdx.x & 31;
  const int lhalf  = lane & 15;
  const int hi     = lane >> 4;          // 0: lanes 0-15, 1: lanes 16-31
  const int m_base = mtile * 128 + wave * 16;

  const __bf16* Abase = tokbf + (size_t)(m_base + lhalf) * Dm;
  const int aoff0 = hi ? 16 : 0;         // A-chunk K offsets within 32-window
  const int aoff1 = hi ? 24 : 8;

  // B staging: thread tid fills the 32B slot [ks][tid>>5][tid&31] of each
  // kstep slab, which maps to codebook column (t*16 + lane&15), K-half lane>>4.
  const int bcol_local = (threadIdx.x >> 5) * 16 + (lhalf);  // 0..127
  const int bkoff      = hi * 16;                            // 0 or 16 elems
  const unsigned ldsBase =
      (unsigned)(unsigned long long)&sB[0] + (unsigned)threadIdx.x * 32u;

  float best_d[8];
  int   best_i[8];
#pragma unroll
  for (int i = 0; i < 8; ++i) { best_d[i] = __builtin_inff(); best_i[i] = 0; }

  const int n_begin = part * (Kc / NSPLIT);
  const int n_end   = n_begin + (Kc / NSPLIT);

#pragma unroll 1
  for (int nc = n_begin; nc < n_end; nc += 128) {
    // ---- issue async panel 0 (kp = 0) into buffer 0 ----
    {
      const __bf16* g0 = cbbf + (size_t)(nc + bcol_local) * Dm + bkoff;
#pragma unroll
      for (int ks = 0; ks < 4; ++ks) {
        unsigned l = ldsBase + (unsigned)(ks * 8192);
        const __bf16* g = g0 + ks * 32;
        asm volatile(
            "global_load_async_to_lds_b128 %0, %1, off\n\t"
            "global_load_async_to_lds_b128 %0, %1, off offset:16"
            :: "v"(l), "v"(g) : "memory");
      }
    }

    v8f acc[8];
    v8f z = {};
#pragma unroll
    for (int t = 0; t < 8; ++t) acc[t] = z;

    int cur = 0;
#pragma unroll 1
    for (int kp = 0; kp < Dm; kp += KPANEL) {
      const bool has_next = (kp + KPANEL) < Dm;

      // Issue next panel's async copies into the other buffer.
      if (has_next) {
        const __bf16* g0 =
            cbbf + (size_t)(nc + bcol_local) * Dm + (kp + KPANEL) + bkoff;
        unsigned lbase = ldsBase + (unsigned)((cur ^ 1) * PANEL_BYTES);
#pragma unroll
        for (int ks = 0; ks < 4; ++ks) {
          unsigned l = lbase + (unsigned)(ks * 8192);
          const __bf16* g = g0 + ks * 32;
          asm volatile(
              "global_load_async_to_lds_b128 %0, %1, off\n\t"
              "global_load_async_to_lds_b128 %0, %1, off offset:16"
              :: "v"(l), "v"(g) : "memory");
        }
      }

      // Hoist this panel's 4 A fragments (latency hides under the wait).
      FragU a[4];
#pragma unroll
      for (int ks = 0; ks < 4; ++ks) {
        a[ks].i[0] = *(const v4i*)(Abase + kp + ks * 32 + aoff0);
        a[ks].i[1] = *(const v4i*)(Abase + kp + ks * 32 + aoff1);
      }

      // Wait for the current panel (8 async ops behind the just-issued 8).
      if (has_next) asm volatile("s_wait_asynccnt 0x8" ::: "memory");
      else          asm volatile("s_wait_asynccnt 0x0" ::: "memory");
      __syncthreads();

      // 32 WMMAs (4 ksteps x 8 tiles): hand-pinned depth-3 LDS pipeline.
      // Slot s = j&3; at step j we issue loads for j+3 and wait on slot j.
      const unsigned sp32 =
          (unsigned)(unsigned long long)&sB[cur * PANEL_BYTES] +
          (unsigned)lane * 32u;
      FragU f[4];
      DS_LOAD_FRAG(f[0], sp32, 0);
      DS_LOAD_FRAG(f[1], sp32, 1024);
      DS_LOAD_FRAG(f[2], sp32, 2048);
#pragma unroll
      for (int j = 0; j < 32; ++j) {
        if (j + 3 < 32) {
          const int jn = j + 3;
          DS_LOAD_FRAG(f[jn & 3], sp32, ((jn >> 3) * 8192 + (jn & 7) * 1024));
        }
        if      (j < 29) DS_WAIT_FRAG(f[j & 3], "0x6");
        else if (j == 29) DS_WAIT_FRAG(f[j & 3], "0x4");
        else if (j == 30) DS_WAIT_FRAG(f[j & 3], "0x2");
        else              DS_WAIT_FRAG(f[j & 3], "0x0");
        const int ks = j >> 3;
        const int t  = j & 7;
        acc[t] = __builtin_amdgcn_wmma_f32_16x16x32_bf16(
            false, a[ks].b, false, f[j & 3].b, (short)0, acc[t], false, false);
      }
      // Final DS_WAIT 0x0 above drained all LDS reads before the swap barrier.
      __syncthreads();
      cur ^= 1;
    }

    // Fold this 128-column chunk into the running per-lane argmin.
#pragma unroll
    for (int t = 0; t < 8; ++t) {
      const int col = nc + t * 16 + lhalf;
      const float es = esq[col];
#pragma unroll
      for (int i = 0; i < 8; ++i) {
        float dist = es - 2.0f * acc[t][i];   // ||x||^2 term is row-constant
        bool take = (dist < best_d[i]) ||
                    (dist == best_d[i] && col < best_i[i]);
        if (take) { best_d[i] = dist; best_i[i] = col; }
      }
    }
  }

  // Reduce across the 16 lanes that share each row (lane = column id).
#pragma unroll
  for (int m = 1; m < 16; m <<= 1) {
#pragma unroll
    for (int i = 0; i < 8; ++i) {
      float od = __shfl_xor(best_d[i], m, 32);
      int   oi = __shfl_xor(best_i[i], m, 32);
      bool take = (od < best_d[i]) || (od == best_d[i] && oi < best_i[i]);
      if (take) { best_d[i] = od; best_i[i] = oi; }
    }
  }

  if (lhalf == 0) {
    const int rbase = m_base + (hi ? 8 : 0);
#pragma unroll
    for (int i = 0; i < 8; ++i) {
      bdist[(size_t)part * Mtot + rbase + i] = best_d[i];
      bidx [(size_t)part * Mtot + rbase + i] = best_i[i];
    }
  }
}

// ---------------------------------------------------------------------------
// Kernel 4: combine partitions, gather f32 codebook rows, commit loss.
// One block per token row.
// ---------------------------------------------------------------------------
__global__ void vq_gather(const float* __restrict__ tok,
                          const float* __restrict__ cb,
                          const float* __restrict__ bdist,
                          const int*   __restrict__ bidx,
                          float* __restrict__ outq,
                          int*   __restrict__ outi,
                          float* __restrict__ outloss) {
  const int row = blockIdx.x;
  const float d0 = bdist[row];
  const float d1 = bdist[Mtot + row];
  const int idx = (d1 < d0) ? bidx[Mtot + row] : bidx[row]; // tie -> lower part
  if (threadIdx.x == 0) outi[row] = idx;

  const float* crow = cb  + (size_t)idx * Dm;
  const float* trow = tok + (size_t)row * Dm;
  float*       qrow = outq + (size_t)row * Dm;

  float s = 0.f;
  for (int d = threadIdx.x; d < Dm; d += blockDim.x) {
    float q = crow[d];
    qrow[d] = q;                         // straight-through forward == gather
    float df = q - trow[d];
    s += df * df;
  }
  __shared__ float red[256];
  red[threadIdx.x] = s;
  __syncthreads();
  for (int o = 128; o > 0; o >>= 1) {
    if ((int)threadIdx.x < o) red[threadIdx.x] += red[threadIdx.x + o];
    __syncthreads();
  }
  if (threadIdx.x == 0) atomicAdd(outloss, red[0] * INV_N);
}

// ---------------------------------------------------------------------------
extern "C" void kernel_launch(void* const* d_in, const int* in_sizes, int n_in,
                              void* d_out, int out_size, void* d_ws, size_t ws_size,
                              hipStream_t stream) {
  const float* token = (const float*)d_in[0];   // [Mtot, Dm]
  const float* cb    = (const float*)d_in[1];   // [Kc, Dm]

  // d_out layout: quantized (Mtot*Dm f32) | indices (Mtot i32) | loss (1 f32)
  float* out_q    = (float*)d_out;
  int*   out_idx  = (int*)((float*)d_out + (size_t)Mtot * Dm);
  float* out_loss = (float*)d_out + (size_t)Mtot * Dm + Mtot;

  // Workspace layout.
  char* ws = (char*)d_ws;
  __bf16* tokbf = (__bf16*)ws;                 ws += (size_t)Mtot * Dm * 2; // 24 MB
  __bf16* cbbf  = (__bf16*)ws;                 ws += (size_t)Kc   * Dm * 2; //  3 MB
  float*  esq   = (float*)ws;                  ws += (size_t)Kc * 4;
  float*  bdist = (float*)ws;                  ws += (size_t)NSPLIT * Mtot * 4;
  int*    bidx  = (int*)ws;

  vq_convert<<<4096, 256, 0, stream>>>(token, cb, tokbf, cbbf);
  vq_esq<<<Kc, 256, 0, stream>>>(cb, esq, out_loss);
  vq_argmin<<<(Mtot / 128) * NSPLIT, 256, 0, stream>>>(tokbf, cbbf, esq,
                                                       bdist, bidx);
  vq_gather<<<Mtot, 256, 0, stream>>>(token, cb, bdist, bidx,
                                      out_q, out_idx, out_loss);
}
